// MultiHeadAttention_32469952758289
// MI455X (gfx1250) — matrix operand order repaired and checked by reading, not compiled, not run on hardware
//
#include <hip/hip_runtime.h>
#include <cstdint>
#include <cstddef>

typedef __attribute__((ext_vector_type(16))) _Float16 v16h;
typedef __attribute__((ext_vector_type(8)))  _Float16 v8h;
typedef __attribute__((ext_vector_type(4)))  _Float16 v4h;
typedef __attribute__((ext_vector_type(8)))  float    v8f;

#define D_MODEL 1024
#define NUM_HEADS 16
#define DK 64
#define BATCH 2
#define SEQ 2048
#define MTOT (BATCH * SEQ)

union FragU { v16h h; uint32_t u[8]; };
union H2U  { _Float16 h[2]; uint32_t u; };

__device__ inline uint32_t pack2_f16(float a, float b) {
  H2U t; t.h[0] = (_Float16)a; t.h[1] = (_Float16)b; return t.u;
}

// xor-16 lane exchange via ds_swizzle (group-of-32: xor=0x10, and=0x1f) -> 0x401f
__device__ inline float swz16f(float x) {
  return __int_as_float(__builtin_amdgcn_ds_swizzle(__float_as_int(x), 0x401f));
}
__device__ inline uint32_t swz16u(uint32_t x) {
  return (uint32_t)__builtin_amdgcn_ds_swizzle((int)x, 0x401f);
}

// A-fragment (16x32 f16, M x K), rows contiguous in K.
// lane(0-15)=M with K=0..15, hi lanes repeat M with K=16..31 (contiguous).
__device__ inline v16h load_fragA_f16(const _Float16* base, int stride, int lane) {
  const int r = lane & 15, hi = lane >> 4;
  const _Float16* p = base + (size_t)r * stride + hi * 16;
  FragU f;
#pragma unroll
  for (int v = 0; v < 4; ++v) {
    f.u[v]     = *reinterpret_cast<const uint32_t*>(p + 2 * v);        // K = hi*16 + 2v,2v+1
    f.u[4 + v] = *reinterpret_cast<const uint32_t*>(p + 8 + 2 * v);    // K = hi*16 + 8 + ...
  }
  return f.h;
}

// B-fragment (32x16 f16, K x N) fed from rows of a row-major [N][K] matrix
// (contiguous k order, same as load_fragA_f16: lanes 0-15 K=0..15, lanes 16-31 K=16..31).
__device__ inline v16h load_fragB_f16(const _Float16* base, int stride, int lane) {
  const int n = lane & 15, kh = lane >> 4;
  const _Float16* p = base + (size_t)n * stride + kh * 16;
  FragU f;
#pragma unroll
  for (int v = 0; v < 8; ++v)
    f.u[v] = *reinterpret_cast<const uint32_t*>(p + 2 * v);
  return f.h;
}

__device__ inline v8f wmma_f16(v16h a, v16h b, v8f c) {
  return __builtin_amdgcn_wmma_f32_16x16x32_f16(false, a, false, b, (short)0, c, false, false);
}

// ---------------------------------------------------------------------------
// f32 -> f16 convert, 4 elements per thread (weights and activations)
// ---------------------------------------------------------------------------
__global__ __launch_bounds__(256) void cvt_f32_f16(const float* __restrict__ src,
                                                   _Float16* __restrict__ dst) {
  const int i = (blockIdx.x * 256 + threadIdx.x) * 4;
  float4 f = *reinterpret_cast<const float4*>(src + i);
  v4h o;
  o[0] = (_Float16)f.x; o[1] = (_Float16)f.y; o[2] = (_Float16)f.z; o[3] = (_Float16)f.w;
  *reinterpret_cast<v4h*>(dst + i) = o;
}

// ---------------------------------------------------------------------------
// GEMM: Out = X @ W^T + bias.  X:[M=4096][K=1024] f16, W:[N=1024][K=1024] f16.
// MODE 0: Out f16 row-major       (Q, K projections)
// MODE 1: Out f16 head-transposed (V projection -> Vt[b][dmodel][seq])
// MODE 2: Out f32 row-major       (output projection -> d_out)
// Block: 256 thr = 8 waves; tile 256(M) x 64(N); wave tile 32(M) x 64(N).
// W tiles streamed to double-buffered LDS with global_load_async_to_lds_b128,
// pipelined on ASYNCcnt (async loads complete in order -> wait <=1).
// ---------------------------------------------------------------------------
template <int MODE>
__global__ __launch_bounds__(256) void gemm_xwT(const _Float16* __restrict__ X,
                                                const _Float16* __restrict__ W,
                                                const float* __restrict__ bias,
                                                void* __restrict__ Outv) {
  __shared__ _Float16 lds[2][64 * 40];  // 2 x (64 N-rows x 32 K, stride 40 halves)
  const int lane = threadIdx.x & 31;
  const int wave = threadIdx.x >> 5;
  const int nblk = blockIdx.x * 64;
  const int mblk = blockIdx.y * 256;
  const int mw   = mblk + wave * 32;

  // per-thread staging coordinates: 16B of the 64x32 W tile
  const int sn = threadIdx.x >> 2;
  const int sk = (threadIdx.x & 3) * 8;
  const _Float16* wsrc = W + (size_t)(nblk + sn) * D_MODEL + sk;
  const uint32_t ldsa0 = (uint32_t)(uintptr_t)(&lds[0][sn * 40 + sk]);
  const uint32_t ldsa1 = (uint32_t)(uintptr_t)(&lds[1][sn * 40 + sk]);

  // prologue: issue tile k0=0 into buffer 0
  asm volatile("global_load_async_to_lds_b128 %0, %1, off"
               :: "v"(ldsa0), "v"(wsrc) : "memory");

  v8f acc[2][4] = {};

  for (int k0 = 0, ib = 0; k0 < D_MODEL; k0 += 32, ib ^= 1) {
    if (k0 + 32 < D_MODEL) {  // issue next tile into the other buffer
      const _Float16* nsrc = wsrc + k0 + 32;
      const uint32_t  ndst = ib ? ldsa0 : ldsa1;
      asm volatile("global_load_async_to_lds_b128 %0, %1, off"
                   :: "v"(ndst), "v"(nsrc) : "memory");
      asm volatile("s_wait_asynccnt 0x1" ::: "memory");  // tile k0 has landed
    } else {
      asm volatile("s_wait_asynccnt 0x0" ::: "memory");
    }
    __syncthreads();  // tile k0 visible to all waves

    const v16h a0 = load_fragA_f16(X + (size_t)mw * D_MODEL + k0, D_MODEL, lane);
    const v16h a1 = load_fragA_f16(X + (size_t)(mw + 16) * D_MODEL + k0, D_MODEL, lane);

#pragma unroll
    for (int nt = 0; nt < 4; ++nt) {
      v16h b = load_fragB_f16(&lds[ib][nt * 16 * 40], 40, lane);
      acc[0][nt] = wmma_f16(a0, b, acc[0][nt]);
      acc[1][nt] = wmma_f16(a1, b, acc[1][nt]);
    }
    __syncthreads();  // all reads of lds[ib] done before tile k0+64 overwrites it
  }

  // Epilogue. C layout: vgpr r, lanes<16 -> (M=r, N=lane); lanes>=16 -> (M=r+8, N=lane-16)
  const int r0 = (lane >> 4) * 8;
#pragma unroll
  for (int half = 0; half < 2; ++half) {
    const int mh = mw + half * 16;
#pragma unroll
    for (int nt = 0; nt < 4; ++nt) {
      const int n  = nblk + nt * 16 + (lane & 15);
      const float bv = bias[n];
      if constexpr (MODE == 0) {
        _Float16* out = (_Float16*)Outv;
#pragma unroll
        for (int r = 0; r < 8; ++r) {
          const int m = mh + r0 + r;
          out[(size_t)m * D_MODEL + n] = (_Float16)(acc[half][nt][r] + bv);
        }
      } else if constexpr (MODE == 1) {
        // Vt[b][n][s], b = token>>11, s = token&2047; 8 consecutive tokens per lane
        _Float16* out = (_Float16*)Outv;
        const int b  = mh >> 11;
        const int s0 = (mh & 2047) + r0;
        v8h vals;
#pragma unroll
        for (int r = 0; r < 8; ++r) vals[r] = (_Float16)(acc[half][nt][r] + bv);
        *reinterpret_cast<v8h*>(out + ((size_t)(b * D_MODEL + n)) * SEQ + s0) = vals;
      } else {
        float* out = (float*)Outv;
#pragma unroll
        for (int r = 0; r < 8; ++r) {
          const int m = mh + r0 + r;
          out[(size_t)m * D_MODEL + n] = acc[half][nt][r] + bv;
        }
      }
    }
  }
}

// ---------------------------------------------------------------------------
// Flash attention: each wave owns TWO 16-row q-tiles (32 q rows), so every
// K / V fragment load is amortized over 2x the WMMAs (halves K/V traffic).
// Computes S^T = K.Q^T (per-q softmax = per-lane ops + one xor16 swizzle),
// then O^T = V^T . P^T accumulated over 64-key blocks, online softmax.
// Scale folded into Q (f16 x0.125 is exact).
// Qh,Kh: [b][s][h*64+d]; Vt: [b][h*64+d][s]; Ao like Qh.
// ---------------------------------------------------------------------------
#define FA_WAVES 4
#define KB 64
__global__ __launch_bounds__(FA_WAVES * 32) void flash_attn(
    const _Float16* __restrict__ Qh, const _Float16* __restrict__ Kh,
    const _Float16* __restrict__ Vt, _Float16* __restrict__ Ao) {
  const int lane = threadIdx.x & 31;
  const int wave = threadIdx.x >> 5;
  const int q0 = (blockIdx.x * FA_WAVES + wave) * 32;
  const int h  = blockIdx.y;
  const int b  = blockIdx.z;
  const int kh = lane >> 4;

  // two q-tiles: j=0 -> rows q0..q0+15, j=1 -> rows q0+16..q0+31
  v16h bq[2][2];
  const _Float16 qs = (_Float16)0.125f;  // 1/sqrt(64), exact in f16
#pragma unroll
  for (int j = 0; j < 2; ++j) {
    const _Float16* Qbase = Qh + ((size_t)(b * SEQ + q0 + 16 * j)) * D_MODEL + h * DK;
    bq[j][0] = load_fragB_f16(Qbase, D_MODEL, lane);
    bq[j][1] = load_fragB_f16(Qbase + 32, D_MODEL, lane);
#pragma unroll
    for (int i = 0; i < 16; ++i) { bq[j][0][i] *= qs; bq[j][1][i] *= qs; }
  }

  v8f o[2][4] = {};
  float m[2] = {-3.0e38f, -3.0e38f}, l[2] = {0.0f, 0.0f};

  for (int kb = 0; kb < SEQ; kb += KB) {
    const _Float16* Kbase = Kh + ((size_t)(b * SEQ + kb)) * D_MODEL + h * DK;
    const _Float16* Vbase = Vt + ((size_t)(b * D_MODEL + h * DK)) * SEQ + kb;
    if (kb + KB < SEQ) {  // uniform branch; prefetch next K/V blocks
      __builtin_prefetch(Kbase + (size_t)KB * D_MODEL, 0, 1);
      __builtin_prefetch(Vbase + KB, 0, 1);
    }

    // scores: S^T tiles, keys (kb + 16t .. +15) x q, shared K frags for both j
    v8f st[2][4];
    const v8f z = {};
#pragma unroll
    for (int t = 0; t < 4; ++t) {
      const v16h a0 = load_fragA_f16(Kbase + (size_t)(16 * t) * D_MODEL, D_MODEL, lane);
      const v16h a1 = load_fragA_f16(Kbase + (size_t)(16 * t) * D_MODEL + 32, D_MODEL, lane);
#pragma unroll
      for (int j = 0; j < 2; ++j) {
        st[j][t] = wmma_f16(a0, bq[j][0], z);
        st[j][t] = wmma_f16(a1, bq[j][1], st[j][t]);
      }
    }

    // online softmax + P^T fragment build, independently per q-tile
    float alpha[2];
    FragU pf[2][2];
#pragma unroll
    for (int j = 0; j < 2; ++j) {
      float mb = -3.0e38f;
#pragma unroll
      for (int t = 0; t < 4; ++t)
#pragma unroll
        for (int i = 0; i < 8; ++i) mb = fmaxf(mb, st[j][t][i]);
      mb = fmaxf(mb, swz16f(mb));
      const float mn = fmaxf(m[j], mb);
      alpha[j] = __expf(m[j] - mn);

      float ls = 0.0f;
      uint32_t up[4][4];
#pragma unroll
      for (int t = 0; t < 4; ++t)
#pragma unroll
        for (int v = 0; v < 4; ++v) {
          const float p0 = __expf(st[j][t][2 * v] - mn);
          const float p1 = __expf(st[j][t][2 * v + 1] - mn);
          ls += p0 + p1;
          up[t][v] = pack2_f16(p0, p1);
        }
      ls += swz16f(ls);
      l[j] = l[j] * alpha[j] + ls;
      m[j] = mn;

      // pf[j][0] = keys kb+0..31 (St0,St1), pf[j][1] = keys kb+32..63 (St2,St3).
      // Low lanes need the partner half-wave's St(2t) keys; high lanes the St(2t+1).
#pragma unroll
      for (int v = 0; v < 4; ++v) {
        const uint32_t x0 = swz16u(up[0][v]);
        const uint32_t x1 = swz16u(up[1][v]);
        pf[j][0].u[v]     = kh ? x1 : up[0][v];
        pf[j][0].u[4 + v] = kh ? up[1][v] : x0;
        const uint32_t x2 = swz16u(up[2][v]);
        const uint32_t x3 = swz16u(up[3][v]);
        pf[j][1].u[v]     = kh ? x3 : up[2][v];
        pf[j][1].u[4 + v] = kh ? up[3][v] : x2;
      }
    }

#pragma unroll
    for (int j = 0; j < 2; ++j)
#pragma unroll
      for (int t = 0; t < 4; ++t)
#pragma unroll
        for (int i = 0; i < 8; ++i) o[j][t][i] *= alpha[j];

    // O^T += V^T . P^T   (V^T frags contiguous rows of Vt, shared for both j)
#pragma unroll
    for (int t = 0; t < 4; ++t) {
      const v16h av0 = load_fragA_f16(Vbase + (size_t)(16 * t) * SEQ, SEQ, lane);
      const v16h av1 = load_fragA_f16(Vbase + (size_t)(16 * t) * SEQ + 32, SEQ, lane);
#pragma unroll
      for (int j = 0; j < 2; ++j) {
        o[j][t] = wmma_f16(av0, pf[j][0].h, o[j][t]);
        o[j][t] = wmma_f16(av1, pf[j][1].h, o[j][t]);
      }
    }
  }

  // Epilogue: O^T C layout: lane -> q = qbase + (lane&15); vgpr r -> d = t*16 + kh*8 + r
#pragma unroll
  for (int j = 0; j < 2; ++j) {
    const float inv = 1.0f / l[j];
    const int q = q0 + 16 * j + (lane & 15);
    _Float16* out = Ao + ((size_t)(b * SEQ + q)) * D_MODEL + h * DK + kh * 8;
#pragma unroll
    for (int t = 0; t < 4; ++t) {
      v8h vals;
#pragma unroll
      for (int r = 0; r < 8; ++r) vals[r] = (_Float16)(o[j][t][r] * inv);
      *reinterpret_cast<v8h*>(out + t * 16) = vals;
    }
  }
}

// ---------------------------------------------------------------------------
extern "C" void kernel_launch(void* const* d_in, const int* in_sizes, int n_in,
                              void* d_out, int out_size, void* d_ws, size_t ws_size,
                              hipStream_t stream) {
  const float* q   = (const float*)d_in[0];
  const float* k   = (const float*)d_in[1];
  const float* v   = (const float*)d_in[2];
  const float* Wq  = (const float*)d_in[3];
  const float* bq  = (const float*)d_in[4];
  const float* Wk  = (const float*)d_in[5];
  const float* bk  = (const float*)d_in[6];
  const float* Wv  = (const float*)d_in[7];
  const float* bv  = (const float*)d_in[8];
  const float* Wo  = (const float*)d_in[9];
  const float* bo  = (const float*)d_in[10];

  char* ws = (char*)d_ws;
  const size_t MB = 1024ull * 1024ull;
  _Float16* WqH = (_Float16*)(ws + 0 * MB);
  _Float16* WkH = (_Float16*)(ws + 2 * MB);
  _Float16* WvH = (_Float16*)(ws + 4 * MB);
  _Float16* WoH = (_Float16*)(ws + 6 * MB);
  _Float16* QH  = (_Float16*)(ws + 8 * MB);   // [B][S][1024] projected Q
  _Float16* KH  = (_Float16*)(ws + 16 * MB);  // [B][S][1024] projected K
  _Float16* VT  = (_Float16*)(ws + 24 * MB);  // [B][1024][S] projected V (head-T)
  _Float16* AT  = (_Float16*)(ws + 32 * MB);  // [B][S][1024] attention out
  _Float16* XqH = (_Float16*)(ws + 40 * MB);  // f16 copies of inputs
  _Float16* XkH = (_Float16*)(ws + 48 * MB);
  _Float16* XvH = (_Float16*)(ws + 56 * MB);

  // 1) Convert weights (1M elems) and activations (4M elems) to f16
  const dim3 cw((D_MODEL * D_MODEL) / (256 * 4));
  const dim3 cx((MTOT * D_MODEL) / (256 * 4));
  cvt_f32_f16<<<cw, 256, 0, stream>>>(Wq, WqH);
  cvt_f32_f16<<<cw, 256, 0, stream>>>(Wk, WkH);
  cvt_f32_f16<<<cw, 256, 0, stream>>>(Wv, WvH);
  cvt_f32_f16<<<cw, 256, 0, stream>>>(Wo, WoH);
  cvt_f32_f16<<<cx, 256, 0, stream>>>(q, XqH);
  cvt_f32_f16<<<cx, 256, 0, stream>>>(k, XkH);
  cvt_f32_f16<<<cx, 256, 0, stream>>>(v, XvH);

  // 2) QKV projections (V stored head-transposed)
  const dim3 gg(D_MODEL / 64, MTOT / 256);
  gemm_xwT<0><<<gg, 256, 0, stream>>>(XqH, WqH, bq, QH);
  gemm_xwT<0><<<gg, 256, 0, stream>>>(XkH, WkH, bk, KH);
  gemm_xwT<1><<<gg, 256, 0, stream>>>(XvH, WvH, bv, VT);

  // 3) Flash attention: grid (q-tile pairs, heads, batch)
  flash_attn<<<dim3(SEQ / (32 * FA_WAVES), NUM_HEADS, BATCH), FA_WAVES * 32, 0, stream>>>(
      QH, KH, VT, AT);

  // 4) Output projection -> f32 d_out
  gemm_xwT<2><<<gg, 256, 0, stream>>>(AT, WoH, bo, (float*)d_out);
}
